// CRF_89232240542155
// MI455X (gfx1250) — matrix operand order repaired and checked by reading, not compiled, not run on hardware
//
#include <hip/hip_runtime.h>
#include <math.h>

// CRF forward (partition function) for B=256, T=2048 steps, L=48 labels.
// Phase 1: B*C workgroups each compose their chunk of step matrices in the
//          exp domain (column-rescaled, base-2 bookkeeping) using
//          v_wmma_f32_16x16x32_f16.
// Phase 2: 256 workgroups fold the C chunk matrices into alpha (log domain)
//          and emit the final logsumexp per batch element.

#define B_SZ    256
#define T_FULL  2050
#define T_STEPS 2048
#define L_LBL   48
#define NEG_INF_V (-10000.0f)
#define A_STRIDE 72   // halves; row = 144 B (16B-aligned runs at k%8==0)
#define R_STRIDE 72   // halves; row = 144 B
#define LOG2E_F 1.4426950408889634f
#define LN2_F   0.6931471805599453f

typedef __attribute__((ext_vector_type(16))) _Float16 v16h;
typedef __attribute__((ext_vector_type(8)))  _Float16 v8h;
typedef __attribute__((ext_vector_type(2)))  _Float16 v2h;
typedef __attribute__((ext_vector_type(8)))  float    v8f;
typedef __attribute__((ext_vector_type(4)))  float    v4f;
typedef __attribute__((ext_vector_type(2)))  float    v2f;

__global__ __launch_bounds__(288)
void crf_chunk_kernel(const float* __restrict__ score,
                      const float* __restrict__ trans,
                      float* __restrict__ Pout,   // [B][C][48][48] log-domain
                      int C, int Tc)
{
    __shared__ __align__(16) float     ETf[L_LBL][L_LBL];      // exp(trans)
    __shared__ __align__(16) _Float16  A_s[L_LBL][A_STRIDE];   // step matrix (exp)
    __shared__ __align__(16) _Float16  R_T[L_LBL][R_STRIDE];   // R transposed: [n][k]
    __shared__ float colpart[3][L_LBL];
    __shared__ float lscale2[L_LBL];                           // per-column log2 scale

    const int tid  = threadIdx.x;
    const int lane = tid & 31;
    const int wave = tid >> 5;          // 0..8
    const int mt   = wave / 3;          // output row tile
    const int nt   = wave % 3;          // output col tile
    const int c    = blockIdx.x;        // chunk
    const int b    = blockIdx.y;        // batch element

    // ---- init: exp(trans) (NEG_INF entries -> exact 0), identity R_T, pads
    for (int idx = tid; idx < L_LBL * L_LBL; idx += blockDim.x) {
        ETf[idx / L_LBL][idx % L_LBL] =
            __builtin_amdgcn_exp2f(trans[idx] * LOG2E_F);
    }
    for (int idx = tid; idx < L_LBL * R_STRIDE; idx += blockDim.x) {
        int r = idx / R_STRIDE, q = idx % R_STRIDE;
        R_T[r][q] = (_Float16)0.0f;
        A_s[r][q] = (_Float16)0.0f;     // zero A pads (k>=48) once
    }
    for (int idx = tid; idx < L_LBL; idx += blockDim.x) {
        R_T[idx][idx] = (_Float16)1.0f; // exp-domain identity
        lscale2[idx]  = 0.0f;
    }
    __syncthreads();

    // per-lane fragment geometry (ISA 7.12.2 layouts, 16x16x32 f16)
    const int rowA = mt * 16 + (lane & 15);
    const int nCol = nt * 16 + (lane & 15);
    const int hiA  = (lane < 16) ? 0 : 8;    // A: K-halves start
    const int hiB  = (lane < 16) ? 0 : 8;    // B: K-halves start
    const int mOff = mt * 16 + ((lane < 16) ? 0 : 8);

    // A_s generation geometry: 288 threads cover 12 rows x 24 pairs per pass
    const int jb = tid / 24;            // row base 0..11 (rows jb + 12*pass)
    const int kk = (tid % 24) * 2;      // even column

    v8f acc;
    #pragma unroll
    for (int v = 0; v < 8; ++v) acc[v] = 0.0f;
    float invLast = 1.0f;

    const float* sp = score + ((size_t)b * T_FULL + (size_t)(c * Tc + 1)) * L_LBL;

    // software-pipelined private logit loads for step 0 (4 rows per thread)
    float sc[4];
    #pragma unroll
    for (int p = 0; p < 4; ++p) sc[p] = sp[jb + 12 * p];

    for (int step = 0; step < Tc; ++step) {
        // A_t = diag(exp(logit)) * exp(trans); 4 passes, 2 halves per store.
        // Each thread computes its own exp(logit) -> no LDS broadcast barrier.
        #pragma unroll
        for (int pass = 0; pass < 4; ++pass) {
            int j = jb + pass * 12;
            float e = __builtin_amdgcn_exp2f(sc[pass] * LOG2E_F);
            v2f ev = *(const v2f*)&ETf[j][kk];
            v2h pk;
            pk[0] = (_Float16)(e * ev[0]);
            pk[1] = (_Float16)(e * ev[1]);
            *(v2h*)&A_s[j][kk] = pk;
        }
        // issue next step's logit loads; overlap A gen + WMMA + reductions
        if (step + 1 < Tc) {
            #pragma unroll
            for (int p = 0; p < 4; ++p) sc[p] = sp[L_LBL + jb + 12 * p];
        }
        if (tid == 0 && step + 2 < Tc)
            __builtin_prefetch(sp + 2 * L_LBL, 0, 3);   // warm L2
        sp += L_LBL;
        __syncthreads();

        // fragment loads: contiguous 16B runs -> ds_load_b128
        v8h aL0 = *(const v8h*)&A_s[rowA][hiA];
        v8h aH0 = *(const v8h*)&A_s[rowA][16 + hiA];
        v8h aL1 = *(const v8h*)&A_s[rowA][32 + hiA];
        v8h aH1 = *(const v8h*)&A_s[rowA][48 + hiA];
        v8h bL0 = *(const v8h*)&R_T[nCol][hiB];
        v8h bH0 = *(const v8h*)&R_T[nCol][16 + hiB];
        v8h bL1 = *(const v8h*)&R_T[nCol][32 + hiB];
        v8h bH1 = *(const v8h*)&R_T[nCol][48 + hiB];
        v16h a0 = __builtin_shufflevector(aL0, aH0, 0,1,2,3,4,5,6,7,8,9,10,11,12,13,14,15);
        v16h a1 = __builtin_shufflevector(aL1, aH1, 0,1,2,3,4,5,6,7,8,9,10,11,12,13,14,15);
        v16h b0 = __builtin_shufflevector(bL0, bH0, 0,1,2,3,4,5,6,7,8,9,10,11,12,13,14,15);
        v16h b1 = __builtin_shufflevector(bL1, bH1, 0,1,2,3,4,5,6,7,8,9,10,11,12,13,14,15);

        // R_new = A_t x R (two K-tiles of 32), f32 accumulate
        #pragma unroll
        for (int v = 0; v < 8; ++v) acc[v] = 0.0f;
        acc = __builtin_amdgcn_wmma_f32_16x16x32_f16(false, a0, false, b0,
                                                     (short)0, acc, false, false);
        acc = __builtin_amdgcn_wmma_f32_16x16x32_f16(false, a1, false, b1,
                                                     (short)0, acc, false, false);

        // per-column max: 8 regs -> lane pair (l, l^16) -> 3 row tiles via LDS
        float m8 = acc[0];
        #pragma unroll
        for (int v = 1; v < 8; ++v) m8 = fmaxf(m8, acc[v]);
        float m16 = fmaxf(m8, __shfl_xor(m8, 16, 32));
        if (lane < 16) colpart[mt][nt * 16 + lane] = m16;
        __syncthreads();

        // every lane derives its own column scale (v_rcp_f32); lscale2 1-writer
        float cm = fmaxf(fmaxf(colpart[0][nCol], colpart[1][nCol]), colpart[2][nCol]);
        cm = fmaxf(cm, 1e-30f);
        float inv = __builtin_amdgcn_rcpf(cm);
        invLast = inv;
        if (tid < L_LBL) {
            float c2 = fmaxf(fmaxf(colpart[0][tid], colpart[1][tid]), colpart[2][tid]);
            lscale2[tid] += __builtin_amdgcn_logf(fmaxf(c2, 1e-30f)); // v_log_f32 = log2
        }

        // write back column-normalized R_T (one ds_store_b128 per lane)
        v8h wnorm;
        #pragma unroll
        for (int v = 0; v < 8; ++v) wnorm[v] = (_Float16)(acc[v] * inv);
        *(v8h*)&R_T[nCol][mOff] = wnorm;
        __syncthreads();
    }

    // emit chunk operator in log domain: ln(R) + ln2*colscale2
    {
        float ls2 = lscale2[nCol];
        #pragma unroll
        for (int v = 0; v < 8; ++v) {
            float val = acc[v] * invLast;
            float lg  = (val >= 1.2e-38f)
                      ? LN2_F * (__builtin_amdgcn_logf(val) + ls2)
                      : -1.0e30f;
            Pout[(((size_t)b * C + c) * L_LBL + (mOff + v)) * L_LBL + nCol] = lg;
        }
    }
}

__global__ __launch_bounds__(64)
void crf_combine_kernel(const float* __restrict__ Pout,
                        const float* __restrict__ trans,
                        float* __restrict__ out, int C)
{
    __shared__ float alpha[L_LBL];
    __shared__ float alpha_new[L_LBL];
    __shared__ float red[L_LBL];

    const int tid = threadIdx.x;
    const int b   = blockIdx.x;

    if (tid < L_LBL) alpha[tid] = (tid == 0) ? 0.0f : NEG_INF_V;
    __syncthreads();

    for (int c = 0; c < C; ++c) {
        if (tid < L_LBL) {
            const v4f* Prow = (const v4f*)(Pout + (((size_t)b * C + c) * L_LBL + tid) * L_LBL);
            float x[L_LBL];
            float m = -3.0e38f;
            #pragma unroll
            for (int q = 0; q < 12; ++q) {
                v4f p = Prow[q];
                #pragma unroll
                for (int r = 0; r < 4; ++r) {
                    float xv = p[r] + alpha[q * 4 + r];
                    x[q * 4 + r] = xv;
                    m = fmaxf(m, xv);
                }
            }
            float s = 0.0f;
            #pragma unroll
            for (int i = 0; i < L_LBL; ++i)
                s += __builtin_amdgcn_exp2f((x[i] - m) * LOG2E_F);
            alpha_new[tid] = m + LN2_F * __builtin_amdgcn_logf(s);
        }
        __syncthreads();
        if (tid < L_LBL) alpha[tid] = alpha_new[tid];
        __syncthreads();
    }

    if (tid < L_LBL) red[tid] = alpha[tid] + trans[1 * L_LBL + tid]; // + trans[EOS]
    __syncthreads();
    if (tid == 0) {
        float m = red[0];
        for (int i = 1; i < L_LBL; ++i) m = fmaxf(m, red[i]);
        float s = 0.0f;
        for (int i = 0; i < L_LBL; ++i)
            s += __builtin_amdgcn_exp2f((red[i] - m) * LOG2E_F);
        out[b] = m + LN2_F * __builtin_amdgcn_logf(s);
    }
}

extern "C" void kernel_launch(void* const* d_in, const int* in_sizes, int n_in,
                              void* d_out, int out_size, void* d_ws, size_t ws_size,
                              hipStream_t stream)
{
    const float* score = (const float*)d_in[0];   // [256][2050][48] f32
    const float* trans = (const float*)d_in[1];   // [48][48] f32
    float* out = (float*)d_out;                   // [256] f32
    float* P   = (float*)d_ws;                    // chunk matrices

    const size_t per_chunk = (size_t)B_SZ * L_LBL * L_LBL * sizeof(float); // 2.36 MB
    int C = 64;
    while (C > 1 && per_chunk * (size_t)C > ws_size) C >>= 1;
    const int Tc = T_STEPS / C;

    crf_chunk_kernel<<<dim3(C, B_SZ), dim3(288), 0, stream>>>(score, trans, P, C, Tc);
    crf_combine_kernel<<<dim3(B_SZ), dim3(64), 0, stream>>>(P, trans, out, C);
}
